// roPEAttentionHead_23716809409080
// MI455X (gfx1250) — matrix operand order repaired and checked by reading, not compiled, not run on hardware
//
#include <hip/hip_runtime.h>
#include <hip/hip_bf16.h>

#define CTX 1024
#define DMODEL 512
#define NB 8

typedef _Float16 v16h __attribute__((ext_vector_type(16)));
typedef _Float16 v8h  __attribute__((ext_vector_type(8)));
typedef float    v8f  __attribute__((ext_vector_type(8)));
typedef unsigned int v4u __attribute__((ext_vector_type(4)));
typedef int          v8i __attribute__((ext_vector_type(8)));
typedef int          v4i __attribute__((ext_vector_type(4)));

__device__ __forceinline__ v16h cat16(v8h lo, v8h hi) {
  return __builtin_shufflevector(lo, hi, 0,1,2,3,4,5,6,7,8,9,10,11,12,13,14,15);
}

// 16 f16 from two 8-half contiguous groups (16B-aligned each)
__device__ __forceinline__ v16h load16h(const _Float16* p0, const _Float16* p1) {
  v8h a = *(const v8h*)p0;
  v8h b = *(const v8h*)p1;
  return cat16(a, b);
}

// ---------------------------------------------------------------------------
// Kernel 0: bulk f32 -> f16 conversion (x and the three weight matrices).
// 8 elements per thread, fully vectorized, one-shot so the GEMM hot loops
// never touch v_cvt chains.
// ---------------------------------------------------------------------------
__global__ void __launch_bounds__(256)
cvt_f16_kernel(const float* __restrict__ src, _Float16* __restrict__ dst, int n8) {
  const int i = blockIdx.x * 256 + threadIdx.x;
  if (i >= n8) return;
  const float4 a = ((const float4*)src)[2 * i];
  const float4 b = ((const float4*)src)[2 * i + 1];
  v8h h;
  h[0]=(_Float16)a.x; h[1]=(_Float16)a.y; h[2]=(_Float16)a.z; h[3]=(_Float16)a.w;
  h[4]=(_Float16)b.x; h[5]=(_Float16)b.y; h[6]=(_Float16)b.z; h[7]=(_Float16)b.w;
  *(v8h*)(dst + 8 * (size_t)i) = h;
}

// ---------------------------------------------------------------------------
// Kernel 1: extract the useful 2MB of R (cos/sin diagonals) from the 1GB blob.
// ---------------------------------------------------------------------------
__global__ void __launch_bounds__(256)
rope_tables_kernel(const float* __restrict__ R,
                   float* __restrict__ cosT, float* __restrict__ sinT) {
  const int c = blockIdx.x;          // position
  const int i = threadIdx.x;         // pair index 0..255
  const size_t base = (size_t)c * DMODEL * DMODEL;
  cosT[c * (DMODEL/2) + i] = R[base + (size_t)(2*i)   * DMODEL + 2*i];
  sinT[c * (DMODEL/2) + i] = R[base + (size_t)(2*i+1) * DMODEL + 2*i];
}

// ---------------------------------------------------------------------------
// Kernel 2: q/k/v = x@W^T + b (f16 operands, f32 accum), fused RoPE for q,k.
//   grid (64, 8, 3): 128-row x 64-col tile per block, blockIdx.z selects W.
//   The block's 64KB weight tile (rows n0..n0+63, all K) is staged into
//   dynamic LDS by the Tensor Data Mover (one descriptor, wave 0 issues,
//   s_wait_tensorcnt, barrier), then all 8 waves read B-fragments from LDS.
//   A-fragments stream from pre-converted f16 x, double-buffered over kt.
// ---------------------------------------------------------------------------
__global__ void __launch_bounds__(256)
qkv_rope_kernel(const _Float16* __restrict__ xh,
                const _Float16* __restrict__ whq,
                const _Float16* __restrict__ whk,
                const _Float16* __restrict__ whv,
                const float* __restrict__ bq, const float* __restrict__ bk,
                const float* __restrict__ bv,
                const float* __restrict__ cosT, const float* __restrict__ sinT,
                _Float16* __restrict__ qh, _Float16* __restrict__ kh,
                _Float16* __restrict__ vt) {
  extern __shared__ _Float16 Ws[];             // 64 rows x 512 halves = 64KB

  const int mat  = blockIdx.z;                 // 0=q 1=k 2=v (uniform)
  const _Float16* wh = (mat == 0) ? whq : (mat == 1) ? whk : whv;
  const float*    Bs = (mat == 0) ? bq  : (mat == 1) ? bk  : bv;

  const int wid  = threadIdx.x >> 5;           // wave 0..7
  const int lane = threadIdx.x & 31;
  const int l15  = lane & 15;
  const int hiA  = (lane < 16) ? 0 : 8;        // A-frag K base within 32
  const int hiB  = (lane < 16) ? 0 : 16;       // unused; B-frag K base is hiA
  const int hi8  = (lane < 16) ? 0 : 8;        // C-frag row offset
  const int rowBase = blockIdx.x * 128 + wid * 16;
  const int n0      = blockIdx.y * 64;
  const int mA      = rowBase + l15;           // A-frag row for this lane

  // ---- stage W tile into LDS via the Tensor Data Mover --------------------
#if __has_builtin(__builtin_amdgcn_tensor_load_to_lds) && __has_builtin(__builtin_amdgcn_s_wait_tensorcnt)
  if (wid == 0) {
    const unsigned long long ga =
        (unsigned long long)(const void*)(wh + (size_t)n0 * DMODEL);
    const unsigned lds_off = (unsigned)(unsigned long long)(const void*)Ws;
    v4u g0;
    g0[0] = 1u;                                   // count=1, user descriptor
    g0[1] = lds_off;                              // lds_addr
    g0[2] = (unsigned)ga;                         // global_addr[31:0]
    g0[3] = (unsigned)((ga >> 32) & 0x1FFFFFFu) | 0x80000000u; // addr[56:32] | type=2
    v8i g1;
    g1[0] = 0x00010000;   // wg_mask=0, data_size=1 (2B), no barrier/iter/pad
    g1[1] = 0x02000000;   // tensor_dim0 = 512 (low16 in [31:16])
    g1[2] = 0x02000000;   // tensor_dim0 hi=0 | tensor_dim1 = 512 (low16)
    g1[3] = 0x02000000;   // tensor_dim1 hi=0 | tile_dim0 = 512
    g1[4] = 64;           // tile_dim1 = 64, tile_dim2 = 0
    g1[5] = 512;          // tensor_dim0_stride[31:0] = 512
    g1[6] = 0;            // dim0_stride hi | dim1_stride lo
    g1[7] = 0;
    v4i gz4 = {0, 0, 0, 0};
    v8i gz8 = {0, 0, 0, 0, 0, 0, 0, 0};
    __builtin_amdgcn_tensor_load_to_lds(g0, g1, gz4, gz4, gz8, 0);
    __builtin_amdgcn_s_wait_tensorcnt(0);
  }
#else
  {  // fallback: cooperative vectorized copy
    const uint4* srcp = (const uint4*)(wh + (size_t)n0 * DMODEL);
    uint4* dstp = (uint4*)Ws;
    for (int i = threadIdx.x; i < 64 * DMODEL / 8; i += 256) dstp[i] = srcp[i];
  }
#endif
  __syncthreads();

  // ---- 16x64 strip per wave, K-loop over 512 in steps of 32 ---------------
  const _Float16* xp = xh + (size_t)mA * DMODEL;
  v8f acc[4] = {{}, {}, {}, {}};
  v16h a_next = load16h(xp + hiA, xp + hiA + 16);
  for (int kt = 0; kt < 16; ++kt) {
    const v16h a = a_next;
    if (kt < 15)
      a_next = load16h(xp + (kt + 1) * 32 + hiA, xp + (kt + 1) * 32 + hiA + 16);
#pragma unroll
    for (int nf = 0; nf < 4; ++nf) {
      const _Float16* wp = &Ws[(nf * 16 + l15) * DMODEL + kt * 32 + hiA];
      v16h bfr = load16h(wp, wp + 16);
      acc[nf] = __builtin_amdgcn_wmma_f32_16x16x32_f16(
          false, a, false, bfr, (short)0, acc[nf], false, false);
    }
  }

  // ---- epilogue: bias, RoPE (q,k), f16 stores (v transposed) --------------
#pragma unroll
  for (int nf = 0; nf < 4; ++nf) {
    const int n = n0 + nf * 16 + l15;
    const float bias = Bs[n];
#pragma unroll
    for (int r = 0; r < 8; ++r) {
      const int m = rowBase + r + hi8;         // global row (b*CTX + c)
      float val = acc[nf][r] + bias;
      if (mat < 2) {                           // RoPE: rotate pair (2i,2i+1)
        float partner = __shfl_xor(val, 1, 32);    // column parity == lane bit0
        const int c = m & (CTX - 1);
        const int i = n >> 1;
        const float cs = cosT[c * (DMODEL/2) + i];
        const float sn = sinT[c * (DMODEL/2) + i];
        val = (n & 1) ? (sn * partner + cs * val)  // odd:  sin*even + cos*odd
                      : (cs * val - sn * partner); // even: cos*even - sin*odd
      }
      const _Float16 hv = (_Float16)val;
      if (mat == 0)      qh[(size_t)m * DMODEL + n] = hv;
      else if (mat == 1) kh[(size_t)m * DMODEL + n] = hv;
      else {
        const int b = m >> 10, c = m & (CTX - 1);
        vt[((size_t)b * DMODEL + n) * CTX + c] = hv;   // transposed V
      }
    }
  }
}

// ---------------------------------------------------------------------------
// Kernel 3: causal flash attention. Block = 4 waves = 16 query rows.
//   KV blocks of 64: wave w computes S[:, w*16..+16) (16 wmma), online
//   softmax stats shared via LDS, P staged through LDS, wave w accumulates
//   O over its private 128-wide D slice (16 wmma).
// ---------------------------------------------------------------------------
__global__ void __launch_bounds__(128)
flash_attn_kernel(const _Float16* __restrict__ qh,
                  const _Float16* __restrict__ kh,
                  const _Float16* __restrict__ vt,
                  float* __restrict__ out) {
  const int qt = blockIdx.x, b = blockIdx.y;
  const int q0 = qt * 16;
  const int tid = threadIdx.x;
  const int w   = tid >> 5;
  const int lane = tid & 31;
  const int l15 = lane & 15;
  const int hi8 = (lane < 16) ? 0 : 8;
  const int kbA = (lane < 16) ? 0 : 8;
  const int kbB = (lane < 16) ? 0 : 16;

  __shared__ __align__(16) _Float16 Qs[16 * DMODEL];  // 16KB
  __shared__ __align__(16) _Float16 Ps[16 * 64];      // 2KB
  __shared__ float sMax[4][16];
  __shared__ float sSum[4][16];
  __shared__ float rowM[16];
  __shared__ float rowL[16];
  __shared__ float alphaS[16];

  {  // stage Q tile (contiguous 16KB) into LDS
    const uint4* src = (const uint4*)(qh + (size_t)(b * CTX + q0) * DMODEL);
    uint4* dst = (uint4*)Qs;
    for (int i = tid; i < 16 * DMODEL / 8; i += 128) dst[i] = src[i];
  }
  if (tid < 16) { rowM[tid] = -__builtin_inff(); rowL[tid] = 0.0f; }
  v8f o[8] = {{}, {}, {}, {}, {}, {}, {}, {}};
  __syncthreads();

  const float scale = 0.044194173824159216f;  // 1/sqrt(512)
  const int jmax = (q0 + 15) >> 6;
  const int d0 = w * 128;

  for (int j = 0; j <= jmax; ++j) {
    const int kv0 = j * 64;

    // ---- S frag for this wave's 16 KV columns (K-frag double-buffered)
    v8f s = {};
    const int kvrow = kv0 + w * 16 + l15;
    const _Float16* kp = kh + ((size_t)(b * CTX) + kvrow) * DMODEL + kbA;
    v16h bnext = load16h(kp, kp + 16);
#pragma unroll 4
    for (int kt = 0; kt < 16; ++kt) {
      const v16h bf = bnext;
      if (kt < 15) bnext = load16h(kp + (kt + 1) * 32, kp + (kt + 1) * 32 + 16);
      v16h a = load16h(&Qs[l15 * DMODEL + kt * 32 + kbA],
                       &Qs[l15 * DMODEL + kt * 32 + kbA + 16]);
      s = __builtin_amdgcn_wmma_f32_16x16x32_f16(false, a, false, bf,
                                                 (short)0, s, false, false);
    }

    // ---- scale + causal mask + per-wave row max
#pragma unroll
    for (int r = 0; r < 8; ++r) {
      const int q = q0 + r + hi8;
      float v = s[r] * scale;
      if (kvrow > q) v = -__builtin_inff();
      s[r] = v;
    }
#pragma unroll
    for (int r = 0; r < 8; ++r) {
      float rm = s[r];
      rm = fmaxf(rm, __shfl_xor(rm, 1, 32));
      rm = fmaxf(rm, __shfl_xor(rm, 2, 32));
      rm = fmaxf(rm, __shfl_xor(rm, 4, 32));
      rm = fmaxf(rm, __shfl_xor(rm, 8, 32));
      if (l15 == 0) sMax[w][r + hi8] = rm;
    }
    __syncthreads();

    if (tid < 16) {
      const float m0 = rowM[tid];
      float mn = m0;
      for (int w2 = 0; w2 < 4; ++w2) mn = fmaxf(mn, sMax[w2][tid]);
      alphaS[tid] = __expf(m0 - mn);
      rowM[tid] = mn;
    }
    __syncthreads();

    // ---- P = exp(S-m), stage to LDS, per-wave row sums, rescale O
#pragma unroll
    for (int r = 0; r < 8; ++r) {
      const int row = r + hi8;
      float p = __expf(s[r] - rowM[row]);
      s[r] = p;
      Ps[row * 64 + w * 16 + l15] = (_Float16)p;
    }
#pragma unroll
    for (int r = 0; r < 8; ++r) {
      float rs = s[r];
      rs += __shfl_xor(rs, 1, 32);
      rs += __shfl_xor(rs, 2, 32);
      rs += __shfl_xor(rs, 4, 32);
      rs += __shfl_xor(rs, 8, 32);
      if (l15 == 0) sSum[w][r + hi8] = rs;
    }
    float alv[8];
#pragma unroll
    for (int r = 0; r < 8; ++r) alv[r] = alphaS[r + hi8];
#pragma unroll
    for (int nf = 0; nf < 8; ++nf)
#pragma unroll
      for (int r = 0; r < 8; ++r) o[nf][r] *= alv[r];
    __syncthreads();

    if (tid < 16)
      rowL[tid] = rowL[tid] * alphaS[tid] +
                  sSum[0][tid] + sSum[1][tid] + sSum[2][tid] + sSum[3][tid];

    // ---- O += P @ V over this wave's 128-wide D slice
    v16h pA0 = load16h(&Ps[l15 * 64 + kbA],      &Ps[l15 * 64 + kbA + 16]);
    v16h pA1 = load16h(&Ps[l15 * 64 + 32 + kbA], &Ps[l15 * 64 + 32 + kbA + 16]);
#pragma unroll
    for (int nf = 0; nf < 8; ++nf) {
      const int d = d0 + nf * 16 + l15;
      const _Float16* vp = vt + ((size_t)(b * DMODEL + d)) * CTX + kv0 + kbA;
      v16h b0 = load16h(vp,      vp + 16);
      v16h b1 = load16h(vp + 32, vp + 48);
      o[nf] = __builtin_amdgcn_wmma_f32_16x16x32_f16(false, pA0, false, b0,
                                                     (short)0, o[nf], false, false);
      o[nf] = __builtin_amdgcn_wmma_f32_16x16x32_f16(false, pA1, false, b1,
                                                     (short)0, o[nf], false, false);
    }
  }

  __syncthreads();
  float linv[8];
#pragma unroll
  for (int r = 0; r < 8; ++r) linv[r] = 1.0f / rowL[r + hi8];
  float* op = out + ((size_t)b * CTX + q0) * DMODEL;
#pragma unroll
  for (int nf = 0; nf < 8; ++nf) {
    const int d = d0 + nf * 16 + l15;
#pragma unroll
    for (int r = 0; r < 8; ++r)
      op[(size_t)(r + hi8) * DMODEL + d] = o[nf][r] * linv[r];
  }
}

// ---------------------------------------------------------------------------
extern "C" void kernel_launch(void* const* d_in, const int* in_sizes, int n_in,
                              void* d_out, int out_size, void* d_ws, size_t ws_size,
                              hipStream_t stream) {
  const float* x  = (const float*)d_in[0];
  const float* wq = (const float*)d_in[1];
  const float* bq = (const float*)d_in[2];
  const float* wk = (const float*)d_in[3];
  const float* bk = (const float*)d_in[4];
  const float* wv = (const float*)d_in[5];
  const float* bv = (const float*)d_in[6];
  const float* R  = (const float*)d_in[7];
  float* out = (float*)d_out;

  char* ws = (char*)d_ws;
  float*    cosT = (float*)(ws);                       // 1 MB
  float*    sinT = (float*)(ws + (1u << 20));          // 1 MB
  _Float16* qh   = (_Float16*)(ws + (2u  << 20));      // 8 MB
  _Float16* kh   = (_Float16*)(ws + (10u << 20));      // 8 MB
  _Float16* vt   = (_Float16*)(ws + (18u << 20));      // 8 MB (transposed V)
  _Float16* xh   = (_Float16*)(ws + (26u << 20));      // 8 MB (f16 x)
  _Float16* whq  = (_Float16*)(ws + (34u << 20));      // 512 KB
  _Float16* whk  = (_Float16*)(ws + (34u << 20) + (512u << 10));
  _Float16* whv  = (_Float16*)(ws + (35u << 20));

  // one-shot f32->f16 conversions
  const int nx8 = NB * CTX * DMODEL / 8;               // 524288
  const int nw8 = DMODEL * DMODEL / 8;                 // 32768
  cvt_f16_kernel<<<dim3((nx8 + 255) / 256), dim3(256), 0, stream>>>(x, xh, nx8);
  cvt_f16_kernel<<<dim3((nw8 + 255) / 256), dim3(256), 0, stream>>>(wq, whq, nw8);
  cvt_f16_kernel<<<dim3((nw8 + 255) / 256), dim3(256), 0, stream>>>(wk, whk, nw8);
  cvt_f16_kernel<<<dim3((nw8 + 255) / 256), dim3(256), 0, stream>>>(wv, whv, nw8);

  rope_tables_kernel<<<dim3(CTX), dim3(256), 0, stream>>>(R, cosT, sinT);
  qkv_rope_kernel<<<dim3(64, 8, 3), dim3(256), 65536, stream>>>(
      xh, whq, whk, whv, bq, bk, bv, cosT, sinT, qh, kh, vt);
  flash_attn_kernel<<<dim3(CTX / 16, NB), dim3(128), 0, stream>>>(qh, kh, vt, out);
}
